// CrossViewTransformer_24498493456756
// MI455X (gfx1250) — matrix operand order repaired and checked by reading, not compiled, not run on hardware
//
#include <hip/hip_runtime.h>
#include <hip/hip_bf16.h>
#include <cmath>

typedef __attribute__((ext_vector_type(2)))  float    v2f;
typedef __attribute__((ext_vector_type(8)))  float    v8f;
typedef __attribute__((ext_vector_type(2)))  _Float16 v2h;
typedef __attribute__((ext_vector_type(8)))  _Float16 v8h;
typedef __attribute__((ext_vector_type(16))) _Float16 v16h;

#define BB   8
#define CC   64
#define NN   4096
#define PROJ 8

#define LOG2E 1.44269504088896340736f

// lane-half swap (lane ^ 16) without an address VGPR: ds_swizzle SWAPX16
// group-of-32 encoding: offset = xor_mask<<10 | or_mask<<5 | and_mask
__device__ __forceinline__ int swap16_i(int x) {
    return __builtin_amdgcn_ds_swizzle(x, 0x401f);
}
__device__ __forceinline__ float swap16_f(float x) {
    return __builtin_bit_cast(float, swap16_i(__builtin_bit_cast(int, x)));
}

// Build the f16 B-operand (32i x 16j) for one 32-wide i chunk from the two
// 16x16 S^T exp tiles.  Pack adjacent-K pairs to 32-bit words FIRST
// (v_cvt_pk_f16_f32), then one half-swap per word.
//   B VGPR r, lanes 0-15 : {P[2r],   P[2r+1]}   (K = 0..15 over r=0..7)
//            lanes 16-31 : {P[16+2r],P[17+2r]}  (K = 16..31)
__device__ __forceinline__ v16h build_bp(const float* e1, const float* e2, bool hi) {
    union { v16h h; int w[8]; } u;
    int pk1[4], pk2[4];
#pragma unroll
    for (int r = 0; r < 4; ++r) {
        pk1[r] = __builtin_bit_cast(int,
                    __builtin_amdgcn_cvt_pkrtz(e1[2 * r], e1[2 * r + 1]));
        pk2[r] = __builtin_bit_cast(int,
                    __builtin_amdgcn_cvt_pkrtz(e2[2 * r], e2[2 * r + 1]));
    }
#pragma unroll
    for (int r = 0; r < 4; ++r) {
        // VGPR r   : lo lanes own pk1[r] (P[0..7]);  hi lanes take partner pk2[r] (P[16..23])
        u.w[r]     = hi ? swap16_i(pk2[r]) : pk1[r];
        // VGPR r+4 : lo lanes take partner pk1[r] (P[8..15]); hi lanes own pk2[r] (P[24..31])
        u.w[4 + r] = hi ? pk2[r] : swap16_i(pk1[r]);
    }
    return u.h;
}

// ---------------------------------------------------------------------------
// Kernel 1: q/k projections.  q is pre-scaled by log2(e) so the softmax can
// use raw v_exp_f32 (exp2) with no per-element multiply in the hot loop.
// ---------------------------------------------------------------------------
__global__ void cvt_proj_qk(const float* __restrict__ qx, const float* __restrict__ rx,
                            const float* __restrict__ wq, const float* __restrict__ bq,
                            const float* __restrict__ wk, const float* __restrict__ bk,
                            float* __restrict__ qb, float* __restrict__ kb) {
    int t = blockIdx.x * blockDim.x + threadIdx.x;   // t = b*NN + i
    int b = t >> 12;
    int i = t & (NN - 1);
    const float* qxp = qx + (size_t)b * CC * NN + i;
    const float* rxp = rx + (size_t)b * CC * NN + i;
    float qa[PROJ], ka[PROJ];
#pragma unroll
    for (int p = 0; p < PROJ; ++p) { qa[p] = 0.f; ka[p] = 0.f; }
    for (int c = 0; c < CC; ++c) {
        float xq = qxp[(size_t)c * NN];
        float xr = rxp[(size_t)c * NN];
#pragma unroll
        for (int p = 0; p < PROJ; ++p) {
            qa[p] = fmaf(wq[p * CC + c], xq, qa[p]);
            ka[p] = fmaf(wk[p * CC + c], xr, ka[p]);
        }
    }
#pragma unroll
    for (int p = 0; p < PROJ; ++p) {
        qb[((size_t)b * PROJ + p) * NN + i] = (qa[p] + bq[p]) * LOG2E;
        kb[((size_t)b * PROJ + p) * NN + i] = ka[p] + bk[p];
    }
}

// ---------------------------------------------------------------------------
// Kernel 2: v projection, stored as f16 (feeds the WMMA A-operand).
// ---------------------------------------------------------------------------
__global__ void cvt_proj_v(const float* __restrict__ rx,
                           const float* __restrict__ wv, const float* __restrict__ bv,
                           _Float16* __restrict__ vb) {
    int t  = blockIdx.x * blockDim.x + threadIdx.x;  // t = ((b*8+cg)*NN) + i
    int i  = t & (NN - 1);
    int bc = t >> 12;
    int b  = bc >> 3;
    int cg = bc & 7;                                  // channel group of 8
    const float* rxp = rx + (size_t)b * CC * NN + i;
    float acc[8];
#pragma unroll
    for (int u = 0; u < 8; ++u) acc[u] = 0.f;
    for (int c2 = 0; c2 < CC; ++c2) {
        float xv = rxp[(size_t)c2 * NN];
#pragma unroll
        for (int u = 0; u < 8; ++u)
            acc[u] = fmaf(wv[(cg * 8 + u) * CC + c2], xv, acc[u]);
    }
#pragma unroll
    for (int u = 0; u < 8; ++u) {
        int c = cg * 8 + u;
        vb[((size_t)b * CC + c) * NN + i] = (_Float16)(acc[u] + bv[c]);
    }
}

// ---------------------------------------------------------------------------
// Kernel 3: flash-attention pass (base-2 softmax domain), 64-wide i steps so
// the per-step softmax bookkeeping (acc rescale, reductions, corr) is paid
// half as often.
//   S^T[i,j] = sum_p (q*log2e)[p,i] k[p,j]   (V_WMMA_F32_16X16X4_F32, K=8)
//   P = 2^(S^T - m[j]),  online m/l per lane (j == lane&15)
//   acc[c,j] += V[c,i] * P^T[i,j]            (V_WMMA_F32_16X16X32_F16)
//   out[b,c,j] = qx[b,c,j] + acc[c,j] / l[j]
// One wave per 16-wide j tile; 4 waves / 128 threads per block; no divergence.
// ---------------------------------------------------------------------------
__global__ void __launch_bounds__(128)
cvt_attention(const float* __restrict__ qx,
              const float* __restrict__ qb, const float* __restrict__ kb,
              const _Float16* __restrict__ vb, float* __restrict__ out) {
    const int lane  = threadIdx.x & 31;
    const int wave  = threadIdx.x >> 5;
    const int wtile = blockIdx.x * 4 + wave;       // 2048 j-tiles total
    const int b     = wtile >> 8;                  // 256 tiles per batch
    const int j0    = (wtile & 255) << 4;
    const bool hi   = lane >= 16;
    const int  l16  = lane & 15;

    const float*    qbp = qb + (size_t)b * PROJ * NN;
    const float*    kbp = kb + (size_t)b * PROJ * NN;
    const _Float16* vbp = vb + (size_t)b * CC * NN;

    // B-operand of the score WMMA: k[p, j0+n], invariant over the i loop.
    // f32 B 4x16 layout (mirrors A): VGPR0 = rows {0|2}, VGPR1 = rows {1|3}.
    const int jn = j0 + l16;
    v2f bk0, bk1;
    bk0[0] = kbp[(size_t)(hi ? 2 : 0) * NN + jn];
    bk0[1] = kbp[(size_t)(hi ? 3 : 1) * NN + jn];
    bk1[0] = kbp[(size_t)(hi ? 6 : 4) * NN + jn];
    bk1[1] = kbp[(size_t)(hi ? 7 : 5) * NN + jn];

    v8f acc[4] = {};                 // D[c,j] tiles for c = 0..63, f32 accum
    float m = -INFINITY, l = 0.f;

    for (int i0 = 0; i0 < NN; i0 += 64) {
        // ---- scores S^T for four 16-wide i sub-tiles (full fp32, K=8) ----
        v8f st[4];
#pragma unroll
        for (int s = 0; s < 4; ++s) {
            const int im = i0 + s * 16 + l16;
            v2f a0, a1;
            a0[0] = qbp[(size_t)(hi ? 2 : 0) * NN + im];
            a0[1] = qbp[(size_t)(hi ? 3 : 1) * NN + im];
            a1[0] = qbp[(size_t)(hi ? 6 : 4) * NN + im];
            a1[1] = qbp[(size_t)(hi ? 7 : 5) * NN + im];
            v8f z = {};
            z = __builtin_amdgcn_wmma_f32_16x16x4_f32(false, a0, false, bk0,
                                                      (short)0, z, false, false);
            st[s] = __builtin_amdgcn_wmma_f32_16x16x4_f32(false, a1, false, bk1,
                                                          (short)0, z, false, false);
        }

        // ---- online softmax, base-2 domain (stats per-lane: j == lane&15) ----
        float mloc = -INFINITY;
#pragma unroll
        for (int s = 0; s < 4; s += 2)
#pragma unroll
            for (int r = 0; r < 8; ++r) {
                mloc = fmaxf(mloc, st[s][r]);
                mloc = fmaxf(mloc, st[s + 1][r]);
            }
        mloc = fmaxf(mloc, swap16_f(mloc));
        float mnew = fmaxf(m, mloc);
        float corr = __builtin_amdgcn_exp2f(m - mnew);   // 2^(m - m')
        m = mnew;

        float e[4][8];
        float rsum = 0.f;
#pragma unroll
        for (int s = 0; s < 4; ++s)
#pragma unroll
            for (int r = 0; r < 8; ++r) {
                e[s][r] = __builtin_amdgcn_exp2f(st[s][r] - m);
                rsum += e[s][r];
            }
        rsum += swap16_f(rsum);
        l = l * corr + rsum;
#pragma unroll
        for (int t = 0; t < 4; ++t) acc[t] *= corr;      // once per 64 i

        // ---- pack P^T B-operands for both 32-wide i chunks ----
        const v16h bp0 = build_bp(e[0], e[1], hi);       // i = i0    .. i0+31
        const v16h bp1 = build_bp(e[2], e[3], hi);       // i = i0+32 .. i0+63

        // ---- PV: acc[c,j] += V(16c x 32i) * P^T(32i x 16j), two K chunks ----
#pragma unroll
        for (int t = 0; t < 4; ++t) {
            const _Float16* vrow = vbp + (size_t)(t * 16 + l16) * NN;
            // f16 A, k order used here (same as bp0/bp1): lanes0-15 hold K 0..15; lanes16-31 K 16..31
#pragma unroll
            for (int kk = 0; kk < 2; ++kk) {
                const int ib = i0 + kk * 32;
                v8h alo = *(const v8h*)(vrow + ib + (hi ? 16 : 0));
                v8h ahi = *(const v8h*)(vrow + ib + 8 + (hi ? 16 : 0));
                v16h av;
#pragma unroll
                for (int r = 0; r < 8; ++r) { av[r] = alo[r]; av[8 + r] = ahi[r]; }
                acc[t] = __builtin_amdgcn_wmma_f32_16x16x32_f16(
                             false, av, false, (kk ? bp1 : bp0),
                             (short)0, acc[t], false, false);
            }
        }
    }

    // ---- epilogue: z = acc / l, residual add, store ----
    const float invl = 1.0f / l;
    const int j = j0 + l16;
#pragma unroll
    for (int t = 0; t < 4; ++t) {
        const int cbase = t * 16 + (hi ? 8 : 0);
#pragma unroll
        for (int r = 0; r < 8; ++r) {
            size_t o = ((size_t)b * CC + cbase + r) * NN + j;
            out[o] = qx[o] + acc[t][r] * invl;
        }
    }
}

// ---------------------------------------------------------------------------
extern "C" void kernel_launch(void* const* d_in, const int* in_sizes, int n_in,
                              void* d_out, int out_size, void* d_ws, size_t ws_size,
                              hipStream_t stream) {
    const float* qx = (const float*)d_in[0];
    const float* rx = (const float*)d_in[1];
    const float* wq = (const float*)d_in[2];
    const float* bq = (const float*)d_in[3];
    const float* wk = (const float*)d_in[4];
    const float* bk = (const float*)d_in[5];
    const float* wv = (const float*)d_in[6];
    const float* bv = (const float*)d_in[7];
    float* out = (float*)d_out;

    float*    qb = (float*)d_ws;                       // B*8*N f32   (1 MB)
    float*    kb = qb + (size_t)BB * PROJ * NN;        // B*8*N f32   (1 MB)
    _Float16* vb = (_Float16*)(kb + (size_t)BB * PROJ * NN); // B*64*N f16 (4 MB)

    cvt_proj_qk<<<(BB * NN) / 256, 256, 0, stream>>>(qx, rx, wq, bq, wk, bk, qb, kb);
    cvt_proj_v<<<(BB * NN * 8) / 256, 256, 0, stream>>>(rx, wv, bv, vb);
    // 2048 j-tiles, 1 wave each, 4 waves per block
    cvt_attention<<<512, 128, 0, stream>>>(qx, qb, kb, vb, out);
}
